// RecEncoderLayer_9079560864472
// MI455X (gfx1250) — matrix operand order repaired and checked by reading, not compiled, not run on hardware
//
#include <hip/hip_runtime.h>
#include <hip/hip_bf16.h>
#include <math.h>

#define B_   32
#define T_   1024
#define D_   256
#define H_   128
#define DFF_ 1024
#define BT_  (B_ * T_)
#define G4_  512            // 4*H
#define EPS_ 1e-3f

typedef __attribute__((ext_vector_type(16))) __bf16 v16bf;
typedef __attribute__((ext_vector_type(8)))  float  v8f;
typedef unsigned int  u32x4 __attribute__((ext_vector_type(4)));
typedef int           i32x4 __attribute__((ext_vector_type(4)));
typedef int           i32x8 __attribute__((ext_vector_type(8)));
typedef unsigned short u16;
typedef unsigned int   u32;

// ---------- helpers ----------

__device__ __forceinline__ __bf16 f2bf16(float f) { return (__bf16)f; }  // native cvt
__device__ __forceinline__ u16 f2bf(float f) {
  return __builtin_bit_cast(u16, (__bf16)f);
}

__device__ __forceinline__ v8f wmma_bf16(v16bf a, v16bf b, v8f c) {
  return __builtin_amdgcn_wmma_f32_16x16x32_bf16(false, a, false, b, (short)0, c,
                                                 false, false);
}

// load 8 consecutive bf16 (16B aligned) into v[base..base+7]
__device__ __forceinline__ void ld8bf(const u16* p, v16bf& v, int base) {
  union { uint4 q; __bf16 b[8]; } t;
  t.q = *reinterpret_cast<const uint4*>(p);
#pragma unroll
  for (int i = 0; i < 8; ++i) v[base + i] = t.b[i];
}

// A-fragment (16x32, MxK) from row-major bf16 [*, ld]
__device__ __forceinline__ v16bf ldA_bf(const u16* A, int ld, int row0, int k0, int lane) {
  const u16* p = A + (size_t)(row0 + (lane & 15)) * ld + k0 + ((lane >> 4) << 3);
  v16bf v;
  ld8bf(p,      v, 0);
  ld8bf(p + 16, v, 8);
  return v;
}

// B-fragment (32x16, KxN) from N-major weight [N][K] (ld = K)
__device__ __forceinline__ v16bf ldB_bf(const u16* Bt, int ld, int n0, int k0, int lane) {
  const u16* p = Bt + (size_t)(n0 + (lane & 15)) * ld + k0 + ((lane >> 4) << 3);
  v16bf v;
  ld8bf(p,      v, 0);
  ld8bf(p + 16, v, 8);
  return v;
}

__device__ __forceinline__ void cvt4(float4 q, v16bf& v, int base) {
  v[base + 0] = f2bf16(q.x); v[base + 1] = f2bf16(q.y);
  v[base + 2] = f2bf16(q.z); v[base + 3] = f2bf16(q.w);
}

// A-fragment (16x32) from row-major f32 source, converted on the fly
__device__ __forceinline__ v16bf ldA_f32(const float* A, int ld, int row0, int k0, int lane) {
  const float* p = A + (size_t)(row0 + (lane & 15)) * ld + k0 + ((lane >> 4) << 3);
  const float4* q = reinterpret_cast<const float4*>(p);
  v16bf v;
  cvt4(q[0], v, 0);  cvt4(q[1], v, 4);
  cvt4(q[4], v, 8);  cvt4(q[5], v, 12);
  return v;
}

// ---------- Tensor Data Mover (D# per CDNA5 ISA) ----------

// group0: count=1 | lds_addr | global_addr(57b) | type=2
__device__ __forceinline__ u32x4 tdm_g0(u32 lds_addr, const void* gptr) {
  unsigned long long ga = (unsigned long long)gptr;
  u32x4 g;
  g[0] = 1u;                                              // count = 1, user mode
  g[1] = lds_addr;                                        // LDS byte address
  g[2] = (u32)ga;                                         // global_addr[31:0]
  g[3] = (u32)((ga >> 32) & 0x01FFFFFFu) | 0x80000000u;   // addr[56:32] | type=2
  return g;
}
// group1: 2D tile of 4-byte elements, width w, height hgt, row stride (4B units)
__device__ __forceinline__ i32x8 tdm_g1(u32 w, u32 hgt, u32 stride0) {
  i32x8 g = {};
  g[0] = (int)(2u << 16);                                   // data_size = 4B
  g[1] = (int)((w & 0xFFFFu) << 16);                        // tensor_dim0[15:0]
  g[2] = (int)((w >> 16) | ((hgt & 0xFFFFu) << 16));        // dim0 hi | tensor_dim1 lo
  g[3] = (int)((w & 0xFFFFu) << 16);                        // dim1 hi(0) | tile_dim0
  g[4] = (int)(hgt & 0xFFFFu);                              // tile_dim1
  g[5] = (int)stride0;                                      // tensor_dim0_stride lo
  return g;
}

__device__ __forceinline__ void tdm_load(u32x4 g0, i32x8 g1) {
  i32x4 z4 = {};
#if __clang_major__ >= 23
  i32x8 z8 = {};
  __builtin_amdgcn_tensor_load_to_lds(g0, g1, z4, z4, z8, 0);
#else
  __builtin_amdgcn_tensor_load_to_lds(g0, g1, z4, z4, 0);
#endif
}
__device__ __forceinline__ void tdm_store(u32x4 g0, i32x8 g1) {
  i32x4 z4 = {};
#if __clang_major__ >= 23
  i32x8 z8 = {};
  __builtin_amdgcn_tensor_store_from_lds(g0, g1, z4, z4, z8, 0);
#else
  __builtin_amdgcn_tensor_store_from_lds(g0, g1, z4, z4, 0);
#endif
}
__device__ __forceinline__ u32 lds_off(const void* p) {
  return (u32)(unsigned long long)p;    // generic LDS address: offset in low 32 bits
}

// ---------- kernel 1: transpose + convert weights to bf16 [cols][rows] ----------

__global__ void k_t_bf16(const float* __restrict__ src, u16* __restrict__ dst,
                         int rows, int cols) {
  int i = blockIdx.x * blockDim.x + threadIdx.x;
  if (i < rows * cols) {
    int r = i / cols, c = i % cols;
    dst[(size_t)c * rows + r] = f2bf(src[(size_t)r * cols + c]);
  }
}

// ---------- kernel 2: xW = x @ W + b for both directions ----------

__global__ void __launch_bounds__(256)
k_xw(const float* __restrict__ x,
     const u16* __restrict__ WfT, const u16* __restrict__ WbT,
     const float* __restrict__ bf, const float* __restrict__ bb,
     float* __restrict__ xw0, float* __restrict__ xw1) {
  const int dir = blockIdx.y;
  const u16*   Wt   = dir ? WbT : WfT;
  const float* bias = dir ? bb  : bf;
  float*       xw   = dir ? xw1 : xw0;
  const int row0 = blockIdx.x * 32;
  const int lane = threadIdx.x & 31;
  const int wave = threadIdx.x >> 5;

  v8f zero = {};
  v8f acc[2][4];
#pragma unroll
  for (int m = 0; m < 2; ++m)
#pragma unroll
    for (int i = 0; i < 4; ++i) acc[m][i] = zero;

#pragma unroll 1
  for (int kt = 0; kt < 8; ++kt) {
    const int k0 = kt * 32;
    if (kt + 1 < 8)
      __builtin_prefetch(Wt + (size_t)(wave * 64) * D_ + k0 + 32, 0, 1);
    v16bf a0 = ldA_f32(x, D_, row0,      k0, lane);
    v16bf a1 = ldA_f32(x, D_, row0 + 16, k0, lane);
#pragma unroll
    for (int i = 0; i < 4; ++i) {
      v16bf b = ldB_bf(Wt, D_, (wave * 4 + i) * 16, k0, lane);
      acc[0][i] = wmma_bf16(a0, b, acc[0][i]);
      acc[1][i] = wmma_bf16(a1, b, acc[1][i]);
    }
  }
#pragma unroll
  for (int i = 0; i < 4; ++i) {
    const int col = (wave * 4 + i) * 16 + (lane & 15);
    const float bv = bias[col];
#pragma unroll
    for (int m = 0; m < 2; ++m) {
      const int rbase = row0 + m * 16 + ((lane >> 4) << 3);
#pragma unroll
      for (int r = 0; r < 8; ++r)
        xw[(size_t)(rbase + r) * G4_ + col] = acc[m][i][r] + bv;
    }
  }
}

// ---------- kernel 3: persistent LSTM scan with TDM double buffering ----------
// block 1024 (32 waves), grid 2 (one WGP per direction).
// LDS: U^T 128K | xW bufA 64K | xW bufB 64K | h(bf16) 8K | h-stage(f32) 16K = 280K.

__global__ void __launch_bounds__(1024)
k_scan(const float* __restrict__ xw0, const float* __restrict__ xw1,
       const u16* __restrict__ UT0, const u16* __restrict__ UT1,
       float* __restrict__ h0, float* __restrict__ h1,
       float* __restrict__ out_tail) {
  extern __shared__ char smem[];
  u16*   UT_lds = (u16*)smem;                      // [512][128] bf16
  float* bufA   = (float*)(smem + 131072);         // [32][512] f32
  float* bufB   = (float*)(smem + 196608);         // [32][512] f32
  u16*   h_lds  = (u16*)(smem + 262144);           // [32][128] bf16
  float* hstage = (float*)(smem + 270336);         // [32][128] f32

  const int dir = blockIdx.x;
  const float* xw  = dir ? xw1 : xw0;
  const u16*   UTg = dir ? UT1 : UT0;
  float*       hsq = dir ? h1  : h0;

  const int tid = threadIdx.x, lane = tid & 31, wave = tid >> 5;

  for (int i = tid; i < B_ * H_; i += 1024) h_lds[i] = 0;   // h0 = 0

  const int t0 = dir ? (T_ - 1) : 0;
  if (wave == 0) {
    // stage U^T (1D: 32768 x 4B) and the first xW tile (2D: 512 x 32, stride T*512)
    tdm_load(tdm_g0(lds_off(UT_lds), UTg), tdm_g1(32768u, 1u, 32768u));
    tdm_load(tdm_g0(lds_off(bufA), xw + (size_t)t0 * G4_),
             tdm_g1((u32)G4_, (u32)B_, (u32)T_ * G4_));
  }
  float c[4] = {0.f, 0.f, 0.f, 0.f};

  const int n0 = wave * 16;
  int cur = 0;
#pragma unroll 1
  for (int s = 0; s < T_; ++s) {
    const int t = dir ? (T_ - 1 - s) : s;
    if (wave == 0) __builtin_amdgcn_s_wait_tensorcnt(0);    // cur buf + prev store done
    __syncthreads();
    float* zb = cur ? bufB : bufA;
    float* nb = cur ? bufA : bufB;
    if (wave == 0 && s + 1 < T_) {                          // prefetch next xW tile
      const int tn = dir ? (T_ - 2 - s) : (s + 1);
      tdm_load(tdm_g0(lds_off(nb), xw + (size_t)tn * G4_),
               tdm_g1((u32)G4_, (u32)B_, (u32)T_ * G4_));
    }

    // z = h @ U + xW[t]  (accumulate in place into zb)
    v8f acc0 = {}, acc1 = {};
#pragma unroll
    for (int kt = 0; kt < 4; ++kt) {                        // K = 128
      const int k0 = kt * 32;
      v16bf bm = ldB_bf(UT_lds, H_, n0, k0, lane);
      v16bf a0 = ldA_bf(h_lds, H_, 0,  k0, lane);
      v16bf a1 = ldA_bf(h_lds, H_, 16, k0, lane);
      acc0 = wmma_bf16(a0, bm, acc0);
      acc1 = wmma_bf16(a1, bm, acc1);
    }
    {
      const int col = n0 + (lane & 15);
      const int rb  = (lane >> 4) << 3;
#pragma unroll
      for (int r = 0; r < 8; ++r) zb[(rb + r) * G4_ + col] += acc0[r];
#pragma unroll
      for (int r = 0; r < 8; ++r) zb[(16 + rb + r) * G4_ + col] += acc1[r];
    }
    __syncthreads();

    // gates
#pragma unroll
    for (int k = 0; k < 4; ++k) {
      const int idx = tid + k * 1024;
      const int b = idx >> 7, j = idx & (H_ - 1);
      const float zi = zb[b * G4_ + j];
      const float zf = zb[b * G4_ + H_ + j];
      const float zg = zb[b * G4_ + 2 * H_ + j];
      const float zo = zb[b * G4_ + 3 * H_ + j];
      const float ig = 1.f / (1.f + __expf(-zi));
      const float fg = 1.f / (1.f + __expf(-zf));
      const float og = 1.f / (1.f + __expf(-zo));
      c[k] = fg * c[k] + ig * zg;
      const float h = og * c[k];
      h_lds[b * H_ + j]  = f2bf(h);
      hstage[b * H_ + j] = h;
      if (s == T_ - 1) {
        float* tl = out_tail + (size_t)dir * 2 * B_ * H_;
        tl[b * H_ + j]           = h;
        tl[B_ * H_ + b * H_ + j] = c[k];
      }
    }
    __syncthreads();
    if (wave == 0) {
      // async store h rows for time t (2D: 128 x 32, stride T*128)
      tdm_store(tdm_g0(lds_off(hstage), hsq + (size_t)t * H_),
                tdm_g1((u32)H_, (u32)B_, (u32)T_ * H_));
    }
    cur ^= 1;
  }
  if (wave == 0) __builtin_amdgcn_s_wait_tensorcnt(0);
}

// ---------- kernel 4: LN1 over concat[hf|hb] -> y (bf16) ----------

__global__ void __launch_bounds__(256)
k_ln1(const float* __restrict__ hf, const float* __restrict__ hb,
      const float* __restrict__ gamma, const float* __restrict__ beta,
      u16* __restrict__ y) {
  const int lane = threadIdx.x & 31;
  const int wave = threadIdx.x >> 5;
  const int row  = blockIdx.x * 8 + wave;
  const size_t base = (size_t)row * H_;
  float v[8];
#pragma unroll
  for (int k = 0; k < 8; ++k) {
    const int col = k * 32 + lane;
    v[k] = (col < H_) ? hf[base + col] : hb[base + col - H_];
  }
  float s = 0.f;
#pragma unroll
  for (int k = 0; k < 8; ++k) s += v[k];
#pragma unroll
  for (int o = 16; o > 0; o >>= 1) s += __shfl_xor(s, o, 32);
  const float mean = s * (1.f / (2 * H_));
  float vs = 0.f;
#pragma unroll
  for (int k = 0; k < 8; ++k) { const float d = v[k] - mean; vs += d * d; }
#pragma unroll
  for (int o = 16; o > 0; o >>= 1) vs += __shfl_xor(vs, o, 32);
  const float rs = rsqrtf(vs * (1.f / (2 * H_)) + EPS_);
#pragma unroll
  for (int k = 0; k < 8; ++k) {
    const int col = k * 32 + lane;
    y[(size_t)row * (2 * H_) + col] = f2bf((v[k] - mean) * rs * gamma[col] + beta[col]);
  }
}

// ---------- kernel 5: fused FFN (GEMM1 + GELU + GEMM2 + LN2) ----------

__global__ void __launch_bounds__(256)
k_ffn(const u16* __restrict__ y, const u16* __restrict__ W1T, const u16* __restrict__ W2T,
      const float* __restrict__ b1, const float* __restrict__ b2,
      const float* __restrict__ g2, const float* __restrict__ be2,
      float* __restrict__ out) {
  extern __shared__ char smem[];
  u16*   y_lds  = (u16*)smem;                                    // [64][256] bf16
  u16*   h1_lds = (u16*)(smem + 64 * 256 * 2);                   // [64][1024] bf16
  float* o_lds  = (float*)(smem + 64 * 256 * 2 + 64 * DFF_ * 2); // [64][256] f32

  const int row0 = blockIdx.x * 64;
  const int tid = threadIdx.x, lane = tid & 31, wave = tid >> 5;
  v8f zero = {};

  if (wave == 0) {                               // TDM-stage 64 rows of y (32KB)
    tdm_load(tdm_g0(lds_off(y_lds), y + (size_t)row0 * 256),
             tdm_g1(8192u, 1u, 8192u));
    __builtin_amdgcn_s_wait_tensorcnt(0);
  }
  __syncthreads();

  // GEMM1: h1 = gelu(y @ W1 + b1), K=256, N=1024
#pragma unroll 1
  for (int i = 0; i < 8; ++i) {
    const int n0 = (wave * 8 + i) * 16;
    v8f acc[4];
#pragma unroll
    for (int m = 0; m < 4; ++m) acc[m] = zero;
#pragma unroll 1
    for (int kt = 0; kt < 8; ++kt) {
      const int k0 = kt * 32;
      if (kt + 1 < 8)
        __builtin_prefetch(W1T + (size_t)n0 * 256 + k0 + 32, 0, 1);
      v16bf bm = ldB_bf(W1T, 256, n0, k0, lane);
#pragma unroll
      for (int m = 0; m < 4; ++m) {
        v16bf am = ldA_bf(y_lds, 256, m * 16, k0, lane);
        acc[m] = wmma_bf16(am, bm, acc[m]);
      }
    }
    const int col = n0 + (lane & 15);
    const float bv = b1[col];
#pragma unroll
    for (int m = 0; m < 4; ++m) {
      const int rb = m * 16 + ((lane >> 4) << 3);
#pragma unroll
      for (int r = 0; r < 8; ++r) {
        const float xg = acc[m][r] + bv;
        const float gl = 0.5f * xg * (1.f + erff(xg * 0.70710678118f));
        h1_lds[(rb + r) * DFF_ + col] = f2bf(gl);
      }
    }
  }
  __syncthreads();

  // GEMM2: o = h1 @ W2 + b2, K=1024, N=256
  {
    v8f acc[4][2];
#pragma unroll
    for (int m = 0; m < 4; ++m) { acc[m][0] = zero; acc[m][1] = zero; }
#pragma unroll 1
    for (int kt = 0; kt < 32; ++kt) {
      const int k0 = kt * 32;
      if (kt + 1 < 32)
        __builtin_prefetch(W2T + (size_t)(wave * 32) * DFF_ + k0 + 32, 0, 1);
      v16bf a[4];
#pragma unroll
      for (int m = 0; m < 4; ++m) a[m] = ldA_bf(h1_lds, DFF_, m * 16, k0, lane);
#pragma unroll
      for (int j = 0; j < 2; ++j) {
        v16bf bm = ldB_bf(W2T, DFF_, (wave * 2 + j) * 16, k0, lane);
#pragma unroll
        for (int m = 0; m < 4; ++m) acc[m][j] = wmma_bf16(a[m], bm, acc[m][j]);
      }
    }
#pragma unroll
    for (int j = 0; j < 2; ++j) {
      const int col = (wave * 2 + j) * 16 + (lane & 15);
      const float bv = b2[col];
#pragma unroll
      for (int m = 0; m < 4; ++m) {
        const int rb = m * 16 + ((lane >> 4) << 3);
#pragma unroll
        for (int r = 0; r < 8; ++r)
          o_lds[(rb + r) * 256 + col] = acc[m][j][r] + bv;
      }
    }
  }
  __syncthreads();

  // LN2 + final store, 8 rows per wave
#pragma unroll 1
  for (int rr = 0; rr < 8; ++rr) {
    const int rl = wave * 8 + rr;
    float v[8];
#pragma unroll
    for (int k = 0; k < 8; ++k) v[k] = o_lds[rl * 256 + k * 32 + lane];
    float s = 0.f;
#pragma unroll
    for (int k = 0; k < 8; ++k) s += v[k];
#pragma unroll
    for (int o = 16; o > 0; o >>= 1) s += __shfl_xor(s, o, 32);
    const float mean = s * (1.f / 256.f);
    float vs = 0.f;
#pragma unroll
    for (int k = 0; k < 8; ++k) { const float d = v[k] - mean; vs += d * d; }
#pragma unroll
    for (int o = 16; o > 0; o >>= 1) vs += __shfl_xor(vs, o, 32);
    const float rs = rsqrtf(vs * (1.f / 256.f) + EPS_);
#pragma unroll
    for (int k = 0; k < 8; ++k) {
      const int col = k * 32 + lane;
      out[(size_t)(row0 + rl) * 256 + col] = (v[k] - mean) * rs * g2[col] + be2[col];
    }
  }
}

// ---------- host launcher ----------

extern "C" void kernel_launch(void* const* d_in, const int* in_sizes, int n_in,
                              void* d_out, int out_size, void* d_ws, size_t ws_size,
                              hipStream_t stream) {
  (void)in_sizes; (void)n_in; (void)out_size; (void)ws_size;
  const float* x   = (const float*)d_in[0];
  const float* Wf  = (const float*)d_in[1];
  const float* Uf  = (const float*)d_in[2];
  const float* bf  = (const float*)d_in[3];
  const float* Wb  = (const float*)d_in[4];
  const float* Ub  = (const float*)d_in[5];
  const float* bb  = (const float*)d_in[6];
  const float* g1  = (const float*)d_in[7];
  const float* be1 = (const float*)d_in[8];
  const float* W1  = (const float*)d_in[9];
  const float* b1  = (const float*)d_in[10];
  const float* W2  = (const float*)d_in[11];
  const float* b2  = (const float*)d_in[12];
  const float* g2  = (const float*)d_in[13];
  const float* be2 = (const float*)d_in[14];

  char* ws = (char*)d_ws;
  size_t off = 0;
  float* xw0 = (float*)(ws + off); off += (size_t)BT_ * G4_ * 4;
  float* xw1 = (float*)(ws + off); off += (size_t)BT_ * G4_ * 4;
  float* hf  = (float*)(ws + off); off += (size_t)BT_ * H_ * 4;
  float* hb  = (float*)(ws + off); off += (size_t)BT_ * H_ * 4;
  u16*   yb  = (u16*)(ws + off);   off += (size_t)BT_ * 2 * H_ * 2;
  u16*   WfT = (u16*)(ws + off);   off += (size_t)D_ * G4_ * 2;
  u16*   WbT = (u16*)(ws + off);   off += (size_t)D_ * G4_ * 2;
  u16*   UfT = (u16*)(ws + off);   off += (size_t)H_ * G4_ * 2;
  u16*   UbT = (u16*)(ws + off);   off += (size_t)H_ * G4_ * 2;
  u16*   W1T = (u16*)(ws + off);   off += (size_t)(2 * H_) * DFF_ * 2;
  u16*   W2T = (u16*)(ws + off);   off += (size_t)DFF_ * D_ * 2;

  int n;
  n = D_ * G4_;        k_t_bf16<<<(n + 255) / 256, 256, 0, stream>>>(Wf, WfT, D_, G4_);
  n = D_ * G4_;        k_t_bf16<<<(n + 255) / 256, 256, 0, stream>>>(Wb, WbT, D_, G4_);
  n = H_ * G4_;        k_t_bf16<<<(n + 255) / 256, 256, 0, stream>>>(Uf, UfT, H_, G4_);
  n = H_ * G4_;        k_t_bf16<<<(n + 255) / 256, 256, 0, stream>>>(Ub, UbT, H_, G4_);
  n = (2 * H_) * DFF_; k_t_bf16<<<(n + 255) / 256, 256, 0, stream>>>(W1, W1T, 2 * H_, DFF_);
  n = DFF_ * D_;       k_t_bf16<<<(n + 255) / 256, 256, 0, stream>>>(W2, W2T, DFF_, D_);

  k_xw<<<dim3(BT_ / 32, 2), 256, 0, stream>>>(x, WfT, WbT, bf, bb, xw0, xw1);

  const size_t smemC = 131072 + 65536 + 65536 + 8192 + 16384;   // 286720 B
  float* out_tail = (float*)d_out + (size_t)BT_ * D_;
  k_scan<<<2, 1024, smemC, stream>>>(xw0, xw1, UfT, UbT, hf, hb, out_tail);

  k_ln1<<<BT_ / 8, 256, 0, stream>>>(hf, hb, g1, be1, yb);

  const size_t smemE = (size_t)64 * 256 * 2 + (size_t)64 * DFF_ * 2 + (size_t)64 * 256 * 4;
  k_ffn<<<BT_ / 64, 256, smemE, stream>>>(yb, W1T, W2T, b1, b2, g2, be2, (float*)d_out);
}
